// MTGRU_22565758173715
// MI455X (gfx1250) — matrix operand order repaired and checked by reading, not compiled, not run on hardware
//
#include <hip/hip_runtime.h>
#include <stdint.h>

// ---------------- problem constants ----------------
#define BB     32
#define TT     1024
#define UU     64
#define H1C    512      // H1 == H2 == 512
#define G1     1536     // 3*H
#define INTERC 48
#define OUTC   64
#define KC     10
#define NPHC   256
#define NPLC   30
#define NPARSC 895296
// psi offsets
#define OFF1 786432     // Whh end
#define OFF2 787968     // bh end
#define OFF3 861696     // Wih end
#define OFF4 894464     // C end
#define OFF5 895232     // D end

typedef unsigned short u16;
typedef unsigned int   u32;
typedef __attribute__((ext_vector_type(16))) __bf16 v16bf;
typedef __attribute__((ext_vector_type(16))) u16    v16u;
typedef __attribute__((ext_vector_type(8)))  float  v8f;
typedef __attribute__((ext_vector_type(4)))  u32    u32x4;

union BFrag { v16u u; v16bf b; u32x4 q[2]; };

__device__ inline u16 f2bf(float f) {
    u32 u = __float_as_uint(f);
    u32 r = u + 0x7FFFu + ((u >> 16) & 1u);   // RNE
    return (u16)(r >> 16);
}
__device__ inline float sigm(float x) { return 1.f / (1.f + __expf(-x)); }

__device__ inline v8f wmma_bf16(const BFrag& a, const BFrag& b, v8f c) {
    return __builtin_amdgcn_wmma_f32_16x16x32_bf16(false, a.b, false, b.b,
                                                   (short)0, c, false, false);
}

// ---------------- f32 -> bf16 convert ----------------
__global__ void k_cvt_bf16(const float* __restrict__ src, u16* __restrict__ dst, long n) {
    long i = (long)blockIdx.x * blockDim.x + threadIdx.x;
    if (i < n) dst[i] = f2bf(src[i]);
}

// ---------------- batched bf16 transpose: src (R x C) -> dst (C x R) ----------------
__global__ void __launch_bounds__(256) k_transpose_bf16(
    const u16* __restrict__ src, u16* __restrict__ dst,
    int R, int Cc, long sS, long sD)
{
    __shared__ u16 tile[32][33];
    src += (long)blockIdx.z * sS;
    dst += (long)blockIdx.z * sD;
    const int r0 = blockIdx.y * 32, c0 = blockIdx.x * 32;
    const int tx = threadIdx.x & 31, ty = threadIdx.x >> 5;   // 32 x 8
    for (int i = ty; i < 32; i += 8) {
        int r = r0 + i, c = c0 + tx;
        tile[i][tx] = (r < R && c < Cc) ? src[(long)r * Cc + c] : (u16)0;
    }
    __syncthreads();
    for (int i = ty; i < 32; i += 8) {
        int c = c0 + i, r = r0 + tx;
        if (c < Cc && r < R) dst[(long)c * R + r] = tile[tx][i];
    }
}

// ---------------- generic batched WMMA GEMM (B pre-transposed: N x K) ----------------
// C[M,N] = A[M,K]*B[K,N] (+bias1[n])(+bias2[n]); A row-major bf16, BT = (N,K) bf16.
// grid: (ceil(N/64), ceil(M/64), batch), block 128; wave = 1 mtile x 4 ntiles.
// N must be a multiple of 16; out-of-range tiles are clamped (computed, not stored)
// so the k-loop is branch-free and accumulators stay register-pinned.
__global__ void __launch_bounds__(128) k_wmma_gemm(
    const u16* __restrict__ A, const u16* __restrict__ BT,
    const float* __restrict__ bias1, const float* __restrict__ bias2,
    float* __restrict__ Cf, u16* __restrict__ Cb,
    int M, int N, int Kd, int lda, int ldcf, int ldcb,
    long sA, long sB, long sCf, long sCb, long sBias)
{
    const int z = blockIdx.z;
    A  += (long)z * sA;
    BT += (long)z * sB;
    const int lane  = threadIdx.x & 31;
    const int wave  = threadIdx.x >> 5;
    const int mtile = blockIdx.y * 4 + wave;
    const int nbase = blockIdx.x * 4;
    if (mtile * 16 >= M) return;
    const int m      = mtile * 16 + (lane & 15);
    const int nlane  = lane & 15;
    const int khalfA = (lane >> 4) << 3;   // 0 / 8
    const int khalfB = (lane >> 4) << 3;   // 0 / 8
    const int lastnt = (N >> 4) - 1;

    long boff[4];
    #pragma unroll
    for (int j = 0; j < 4; j++) {
        int ntj = nbase + j;
        if (ntj > lastnt) ntj = lastnt;        // clamp: compute-but-don't-store
        boff[j] = (long)(ntj * 16 + nlane) * Kd;
    }
    v8f acc[4];
    #pragma unroll
    for (int j = 0; j < 4; j++) acc[j] = (v8f){0.f,0.f,0.f,0.f,0.f,0.f,0.f,0.f};

    for (int kk = 0; kk < Kd; kk += 32) {
        BFrag a;
        const u16* ap = A + (long)m * lda + kk + khalfA;
        a.q[0] = *(const u32x4*)(ap);
        a.q[1] = *(const u32x4*)(ap + 16);
        #pragma unroll
        for (int j = 0; j < 4; j++) {
            BFrag b;
            const u16* bp = BT + boff[j] + kk + khalfB;
            b.q[0] = *(const u32x4*)(bp);
            b.q[1] = *(const u32x4*)(bp + 16);
            acc[j] = wmma_bf16(a, b, acc[j]);
        }
    }
    #pragma unroll
    for (int j = 0; j < 4; j++) {
        if ((nbase + j) * 16 >= N) continue;
        const int ncol = (nbase + j) * 16 + nlane;
        float bv = 0.f;
        if (bias1) bv += bias1[(long)z * sBias + ncol];
        if (bias2) bv += bias2[ncol];
        #pragma unroll
        for (int r = 0; r < 8; r++) {
            int row = mtile * 16 + r + khalfA;      // lanes 16..31 hold M = 8+r
            float v = acc[j][r] + bv;
            if (Cf) Cf[(long)z * sCf + (long)row * ldcf + ncol] = v;
            if (Cb) Cb[(long)z * sCb + (long)row * ldcb + ncol] = f2bf(v);
        }
    }
}

// ---------------- hypernetwork: small stage ----------------
__global__ void k_hyper_small(const float* __restrict__ mu, const float* __restrict__ sd,
                              const float* __restrict__ eps,
                              const float* __restrict__ pW1, const float* __restrict__ pb1,
                              const float* __restrict__ pW2, const float* __restrict__ pb2,
                              float* __restrict__ a2out)
{
    __shared__ float zsh[BB * KC];
    __shared__ float a1sh[BB * NPHC];
    const int tid = threadIdx.x;
    for (int i = tid; i < BB * KC; i += blockDim.x) zsh[i] = mu[i] + eps[i] * sd[i];
    __syncthreads();
    for (int i = tid; i < BB * NPHC; i += blockDim.x) {
        int b = i / NPHC, n = i % NPHC;
        float acc = pb1[n];
        #pragma unroll
        for (int k = 0; k < KC; k++) acc += zsh[b * KC + k] * pW1[k * NPHC + n];
        a1sh[i] = tanhf(acc);
    }
    __syncthreads();
    for (int i = tid; i < BB * NPLC; i += blockDim.x) {
        int b = i / NPLC, n = i % NPLC;
        float acc = pb2[n];
        for (int k = 0; k < NPHC; k++) acc += a1sh[b * NPHC + k] * pW2[k * NPLC + n];
        a2out[i] = acc;
    }
}

// ---------------- hypernetwork: psi = a2 @ pW3 + pb3, scattered (coalesced) ----------
__global__ void __launch_bounds__(256) k_hyper_big(
    const float* __restrict__ a2, const float* __restrict__ pW3,
    const float* __restrict__ pb3,
    u16* __restrict__ Whhb, float* __restrict__ bh, u16* __restrict__ Wih2p,
    u16* __restrict__ Cmat, float* __restrict__ Dmat, float* __restrict__ dvec)
{
    __shared__ float a2s[BB * NPLC];
    for (int i = threadIdx.x; i < BB * NPLC; i += blockDim.x) a2s[i] = a2[i];
    __syncthreads();
    long n = (long)blockIdx.x * blockDim.x + threadIdx.x;
    if (n >= NPARSC) return;
    float acc[BB];
    const float base = pb3[n];
    #pragma unroll
    for (int b = 0; b < BB; b++) acc[b] = base;
    for (int k = 0; k < NPLC; k++) {
        float w = pW3[(long)k * NPARSC + n];
        #pragma unroll
        for (int b = 0; b < BB; b++) acc[b] += a2s[b * NPLC + k] * w;
    }
    if (n < OFF1) {                              // Whh (B,512,1536) -> bf16 row-major
        for (int b = 0; b < BB; b++) Whhb[(long)b * OFF1 + n] = f2bf(acc[b]);
    } else if (n < OFF2) {                       // bh (B,1536) f32
        long i = n - OFF1;
        for (int b = 0; b < BB; b++) bh[(long)b * G1 + i] = acc[b];
    } else if (n < OFF3) {                       // Wih (B,48,1536) -> bf16, rows padded to 64
        long i = n - OFF2; long row = i / G1, col = i % G1;
        for (int b = 0; b < BB; b++)
            Wih2p[((long)b * 64 + row) * G1 + col] = f2bf(acc[b]);
    } else if (n < OFF4) {                       // C (B,512,64) -> bf16
        long i = n - OFF3;
        for (int b = 0; b < BB; b++) Cmat[(long)b * (H1C * OUTC) + i] = f2bf(acc[b]);
    } else if (n < OFF5) {                       // D (B,48,16) f32
        long i = n - OFF4;
        for (int b = 0; b < BB; b++) Dmat[(long)b * 768 + i] = acc[b];
    } else {                                     // d (B,64) f32
        long i = n - OFF5;
        for (int b = 0; b < BB; b++) dvec[(long)b * OUTC + i] = acc[b];
    }
}

// ---------------- GRU1 recurrence (shared weights, WhhT = 1536 x 512 bf16) ----------
__global__ void __launch_bounds__(384, 1) k_gru1_scan(
    const float* __restrict__ gx1, const u16* __restrict__ Whh1T,
    float* __restrict__ hglob, u16* __restrict__ seq1, unsigned* __restrict__ ctr)
{
    __shared__ __align__(16) u16 hb[BB * H1C];   // h (bf16), full 32x512
    __shared__ float gh[BB * 96];
    const int s    = blockIdx.x;
    const int tid  = threadIdx.x;
    const int lane = tid & 31;
    const int wave = tid >> 5;                   // 0..11
    const int mt   = wave / 6;                   // 0..1
    const int nt   = wave % 6;                   // 0..5
    const int g    = nt >> 1;
    const int ncol = g * H1C + s * 32 + (nt & 1) * 16 + (lane & 15);
    const int khalfA = (lane >> 4) << 3;
    const int khalfB = (lane >> 4) << 3;

    BFrag Bf[16];                                // 512x16 column slab in VGPRs
    #pragma unroll
    for (int kt = 0; kt < 16; kt++) {
        const u16* wp = Whh1T + (long)ncol * H1C + kt * 32 + khalfB;
        Bf[kt].q[0] = *(const u32x4*)(wp);
        Bf[kt].q[1] = *(const u32x4*)(wp + 16);
    }
    for (int i = tid; i < BB * H1C; i += 384) hb[i] = 0;
    __syncthreads();

    const int mrow = mt * 16 + (lane & 15);
    for (int t = 0; t < TT; t++) {
        v8f acc = {0.f, 0.f, 0.f, 0.f, 0.f, 0.f, 0.f, 0.f};
        #pragma unroll
        for (int kt = 0; kt < 16; kt++) {
            BFrag a;
            const u16* hp = hb + mrow * H1C + kt * 32 + khalfA;
            a.q[0] = *(const u32x4*)(hp);
            a.q[1] = *(const u32x4*)(hp + 16);
            acc = wmma_bf16(a, Bf[kt], acc);
        }
        #pragma unroll
        for (int r = 0; r < 8; r++) {
            int row = mt * 16 + r + khalfA;
            gh[row * 96 + nt * 16 + (lane & 15)] = acc[r];
        }
        __syncthreads();
        for (int e = tid; e < BB * 32; e += 384) {
            int jsub = e & 31, b = e >> 5;
            int j = s * 32 + jsub;
            long base = ((long)b * TT + t) * G1 + s * 32 + jsub;
            float xr = gx1[base], xz = gx1[base + H1C], xn = gx1[base + 2 * H1C];
            if (t + 1 < TT) __builtin_prefetch(&gx1[base + G1], 0, 3);
            float hr = gh[b * 96 + jsub], hz = gh[b * 96 + 32 + jsub], hn = gh[b * 96 + 64 + jsub];
            float rg = sigm(xr + hr);
            float zg = sigm(xz + hz);
            float nn = tanhf(xn + rg * hn);
            float hp = hglob[(long)(t & 1) * BB * H1C + (long)b * H1C + j];
            float hv = zg * hp + (1.f - zg) * nn;
            hglob[(long)((t + 1) & 1) * BB * H1C + (long)b * H1C + j] = hv;
            seq1[((long)b * TT + t) * H1C + j] = f2bf(hv);
        }
        __threadfence();
        __syncthreads();
        if (tid == 0) {                          // grid barrier over 16 WGs
            __hip_atomic_fetch_add(ctr, 1u, __ATOMIC_RELEASE, __HIP_MEMORY_SCOPE_AGENT);
            unsigned target = 16u * (unsigned)(t + 1);
            while (__hip_atomic_load(ctr, __ATOMIC_ACQUIRE, __HIP_MEMORY_SCOPE_AGENT) < target)
                __builtin_amdgcn_s_sleep(1);
        }
        __syncthreads();
        float* src = hglob + (long)((t + 1) & 1) * BB * H1C;
        for (int i = tid; i < BB * H1C; i += 384) {
            float v = __hip_atomic_load(src + i, __ATOMIC_RELAXED, __HIP_MEMORY_SCOPE_AGENT);
            hb[i] = f2bf(v);
        }
        __syncthreads();
    }
}

// ---------------- decoder recurrence (WhhT[b] = 1536 x 512 bf16) ----------------
__global__ void __launch_bounds__(192, 1) k_dec_scan(
    const float* __restrict__ gx2, const u16* __restrict__ WhhT,
    float* __restrict__ hdec, u16* __restrict__ decb,
    float* __restrict__ states, unsigned* __restrict__ ctr)
{
    __shared__ float hfl[H1C];
    __shared__ __align__(16) u16 hbl[H1C];
    __shared__ float ghl[96];
    const int b    = blockIdx.x >> 4;
    const int s    = blockIdx.x & 15;
    const int tid  = threadIdx.x;
    const int lane = tid & 31;
    const int nt   = tid >> 5;                   // 0..5
    const int g    = nt >> 1;
    const int ncol = g * H1C + s * 32 + (nt & 1) * 16 + (lane & 15);
    const int khalfA = (lane >> 4) << 3;
    const int khalfB = (lane >> 4) << 3;
    const u16* WbT = WhhT + (long)b * OFF1;

    BFrag Bf[16];
    #pragma unroll
    for (int kt = 0; kt < 16; kt++) {
        const u16* wp = WbT + (long)ncol * H1C + kt * 32 + khalfB;
        Bf[kt].q[0] = *(const u32x4*)(wp);
        Bf[kt].q[1] = *(const u32x4*)(wp + 16);
    }
    for (int i = tid; i < H1C; i += 192) { hfl[i] = 0.f; hbl[i] = 0; }
    __syncthreads();

    const int mrow = lane & 15;
    const u32x4 zero4 = {0u, 0u, 0u, 0u};
    for (int t = 0; t < TT; t++) {
        v8f acc = {0.f, 0.f, 0.f, 0.f, 0.f, 0.f, 0.f, 0.f};
        #pragma unroll
        for (int kt = 0; kt < 16; kt++) {
            const u16* hp = hbl + kt * 32 + khalfA;
            u32x4 c0 = *(const u32x4*)(hp);
            u32x4 c1 = *(const u32x4*)(hp + 16);
            BFrag a;
            a.q[0] = (mrow == 0) ? c0 : zero4;   // only M=0 row valid
            a.q[1] = (mrow == 0) ? c1 : zero4;
            acc = wmma_bf16(a, Bf[kt], acc);
        }
        if (lane < 16) ghl[nt * 16 + lane] = acc[0];   // VGPR0 / lanes0-15 == M=0
        __syncthreads();
        if (tid < 32) {
            int jsub = tid, j = s * 32 + jsub;
            long base = ((long)b * TT + t) * G1 + s * 32 + jsub;
            float xr = gx2[base], xz = gx2[base + H1C], xn = gx2[base + 2 * H1C];
            if (t + 1 < TT) __builtin_prefetch(&gx2[base + G1], 0, 3);
            float hr = ghl[jsub], hz = ghl[32 + jsub], hn = ghl[64 + jsub];
            float rg = sigm(xr + hr);
            float zg = sigm(xz + hz);
            float et = tanhf(xn + rg * hn);
            float hp = hfl[j];
            float hv = zg * hp + (1.f - zg) * et;
            hdec[((long)b * 2 + ((t + 1) & 1)) * H1C + j] = hv;
            decb[((long)b * TT + t) * H1C + j] = f2bf(hv);
            if (t == TT - 1) states[(long)b * H1C + j] = hv;
        }
        __threadfence();
        __syncthreads();
        if (tid == 0) {                          // barrier over this batch's 16 WGs
            __hip_atomic_fetch_add(ctr + b, 1u, __ATOMIC_RELEASE, __HIP_MEMORY_SCOPE_AGENT);
            unsigned target = 16u * (unsigned)(t + 1);
            while (__hip_atomic_load(ctr + b, __ATOMIC_ACQUIRE, __HIP_MEMORY_SCOPE_AGENT) < target)
                __builtin_amdgcn_s_sleep(1);
        }
        __syncthreads();
        float* src = hdec + ((long)b * 2 + ((t + 1) & 1)) * H1C;
        for (int i = tid; i < H1C; i += 192) {
            float v = __hip_atomic_load(src + i, __ATOMIC_RELAXED, __HIP_MEMORY_SCOPE_AGENT);
            hfl[i] = v; hbl[i] = f2bf(v);
        }
        __syncthreads();
    }
}

// ---------------- residual: yh += [inter, inter@D] ----------------
__global__ void k_resid(const float* __restrict__ inter, const float* __restrict__ Dmat,
                        float* __restrict__ yh)
{
    long idx = (long)blockIdx.x * blockDim.x + threadIdx.x;
    if (idx >= (long)BB * TT * OUTC) return;
    int o = (int)(idx & 63);
    long bt = idx >> 6;
    float add;
    if (o < INTERC) {
        add = inter[bt * INTERC + o];
    } else {
        int b = (int)(bt / TT);
        const float* Db = Dmat + (long)b * 768 + (o - INTERC);
        const float* ip = inter + bt * INTERC;
        float acc = 0.f;
        #pragma unroll
        for (int i = 0; i < INTERC; i++) acc += ip[i] * Db[i * 16];
        add = acc;
    }
    yh[idx] += add;
}

// ---------------- host launch ----------------
extern "C" void kernel_launch(void* const* d_in, const int* in_sizes, int n_in,
                              void* d_out, int out_size, void* d_ws, size_t ws_size,
                              hipStream_t stream)
{
    (void)in_sizes; (void)n_in; (void)out_size; (void)ws_size;
    const float* inp  = (const float*)d_in[0];
    const float* mu   = (const float*)d_in[1];
    const float* sd   = (const float*)d_in[2];
    const float* eps  = (const float*)d_in[3];
    const float* Wih1 = (const float*)d_in[4];
    const float* Whh1 = (const float*)d_in[5];
    const float* bih1 = (const float*)d_in[6];
    const float* bhh1 = (const float*)d_in[7];
    const float* C1w  = (const float*)d_in[8];
    const float* C1b  = (const float*)d_in[9];
    const float* pW1  = (const float*)d_in[10];
    const float* pb1  = (const float*)d_in[11];
    const float* pW2  = (const float*)d_in[12];
    const float* pb2  = (const float*)d_in[13];
    const float* pW3  = (const float*)d_in[14];
    const float* pb3  = (const float*)d_in[15];
    float* out    = (float*)d_out;
    float* states = out + (size_t)BB * TT * OUTC;

    char* p = (char*)d_ws;
    auto take = [&](size_t bytes) -> void* {
        void* r = p; p += (bytes + 255) & ~(size_t)255; return r;
    };
    float* gx1    = (float*)take((size_t)BB * TT * G1 * 4);
    float* gx2    = (float*)take((size_t)BB * TT * G1 * 4);
    u16*   Whhb   = (u16*)  take((size_t)BB * H1C * G1 * 2);   // row-major (hyper output)
    u16*   WhhbT  = (u16*)  take((size_t)BB * H1C * G1 * 2);   // (1536,512) per batch
    u16*   seq1b  = (u16*)  take((size_t)BB * TT * H1C * 2);
    u16*   decb   = (u16*)  take((size_t)BB * TT * H1C * 2);
    float* interf = (float*)take((size_t)BB * TT * INTERC * 4);
    u16*   interp = (u16*)  take((size_t)BB * TT * 64 * 2);
    u16*   Wih2p  = (u16*)  take((size_t)BB * 64 * G1 * 2);
    u16*   Wih2pT = (u16*)  take((size_t)BB * 64 * G1 * 2);    // (1536,64) per batch
    u16*   Cmatb  = (u16*)  take((size_t)BB * H1C * OUTC * 2);
    u16*   CmatbT = (u16*)  take((size_t)BB * H1C * OUTC * 2); // (64,512) per batch
    float* bh     = (float*)take((size_t)BB * G1 * 4);
    float* Dmat   = (float*)take((size_t)BB * 768 * 4);
    float* dvec   = (float*)take((size_t)BB * OUTC * 4);
    u16*   inb    = (u16*)  take((size_t)BB * TT * UU * 2);
    u16*   Wih1b  = (u16*)  take((size_t)UU * G1 * 2);
    u16*   Wih1T  = (u16*)  take((size_t)UU * G1 * 2);         // (1536,64)
    u16*   Whh1b  = (u16*)  take((size_t)H1C * G1 * 2);
    u16*   Whh1T  = (u16*)  take((size_t)H1C * G1 * 2);        // (1536,512)
    u16*   C1wb   = (u16*)  take((size_t)H1C * INTERC * 2);
    u16*   C1wT   = (u16*)  take((size_t)H1C * INTERC * 2);    // (48,512)
    float* a2     = (float*)take((size_t)BB * NPLC * 4);
    float* hglob  = (float*)take((size_t)2 * BB * H1C * 4);
    float* hdec   = (float*)take((size_t)BB * 2 * H1C * 4);
    unsigned* ctrs = (unsigned*)take(256);

    // zero: barrier counters, padded buffers, h ping-pong state
    hipMemsetAsync(ctrs, 0, 256, stream);
    hipMemsetAsync(Wih2p, 0, (size_t)BB * 64 * G1 * 2, stream);
    hipMemsetAsync(interp, 0, (size_t)BB * TT * 64 * 2, stream);
    hipMemsetAsync(hglob, 0, (size_t)2 * BB * H1C * 4, stream);
    hipMemsetAsync(hdec, 0, (size_t)BB * 2 * H1C * 4, stream);

    // bf16 conversions of static operands
    {
        long n;
        n = (long)BB * TT * UU;  k_cvt_bf16<<<(n + 255) / 256, 256, 0, stream>>>(inp,  inb,   n);
        n = (long)UU * G1;       k_cvt_bf16<<<(n + 255) / 256, 256, 0, stream>>>(Wih1, Wih1b, n);
        n = (long)H1C * G1;      k_cvt_bf16<<<(n + 255) / 256, 256, 0, stream>>>(Whh1, Whh1b, n);
        n = (long)H1C * INTERC;  k_cvt_bf16<<<(n + 255) / 256, 256, 0, stream>>>(C1w,  C1wb,  n);
    }
    // transposed (N x K) copies for WMMA B operands
    k_transpose_bf16<<<dim3(48, 2, 1),  256, 0, stream>>>(Wih1b, Wih1T, UU,  G1, 0, 0);
    k_transpose_bf16<<<dim3(48, 16, 1), 256, 0, stream>>>(Whh1b, Whh1T, H1C, G1, 0, 0);
    k_transpose_bf16<<<dim3(2, 16, 1),  256, 0, stream>>>(C1wb,  C1wT,  H1C, INTERC, 0, 0);

    // gx1 = inputs @ Wih1 + (bih1 + bhh1)   [biases folded; gh needs no bias]
    k_wmma_gemm<<<dim3(G1 / 64, (BB * TT) / 64, 1), 128, 0, stream>>>(
        inb, Wih1T, bih1, bhh1, gx1, nullptr,
        BB * TT, G1, UU, UU, G1, 0, 0, 0, 0, 0, 0);

    // hypernetwork
    k_hyper_small<<<1, 256, 0, stream>>>(mu, sd, eps, pW1, pb1, pW2, pb2, a2);
    k_hyper_big<<<(NPARSC + 255) / 256, 256, 0, stream>>>(
        a2, pW3, pb3, Whhb, bh, Wih2p, Cmatb, Dmat, dvec);
    k_transpose_bf16<<<dim3(48, 16, BB), 256, 0, stream>>>(
        Whhb, WhhbT, H1C, G1, (long)H1C * G1, (long)H1C * G1);
    k_transpose_bf16<<<dim3(48, 2, BB), 256, 0, stream>>>(
        Wih2p, Wih2pT, 64, G1, (long)64 * G1, (long)64 * G1);
    k_transpose_bf16<<<dim3(2, 16, BB), 256, 0, stream>>>(
        Cmatb, CmatbT, H1C, OUTC, (long)H1C * OUTC, (long)H1C * OUTC);

    // GRU1 scan -> seq1 (bf16)
    k_gru1_scan<<<16, 384, 0, stream>>>(gx1, Whh1T, hglob, seq1b, ctrs);

    // inter = seq1 @ C1w + C1b  (f32 + zero-padded bf16)
    k_wmma_gemm<<<dim3(1, (BB * TT) / 64, 1), 128, 0, stream>>>(
        seq1b, C1wT, C1b, nullptr, interf, interp,
        BB * TT, INTERC, H1C, H1C, INTERC, 64, 0, 0, 0, 0, 0);

    // gx2[b] = inter_pad[b] @ Wih_pad[b] + bh[b]
    k_wmma_gemm<<<dim3(G1 / 64, TT / 64, BB), 128, 0, stream>>>(
        interp, Wih2pT, bh, nullptr, gx2, nullptr,
        TT, G1, 64, 64, G1, 0,
        (long)TT * 64, (long)64 * G1, (long)TT * G1, 0, G1);

    // decoder scan -> dec (bf16) + states (f32)
    k_dec_scan<<<BB * 16, 192, 0, stream>>>(gx2, WhhbT, hdec, decb, states, ctrs + 32);

    // yhats = dec[b] @ C[b] + d[b]
    k_wmma_gemm<<<dim3(1, TT / 64, BB), 128, 0, stream>>>(
        decb, CmatbT, dvec, nullptr, out, nullptr,
        TT, OUTC, H1C, H1C, OUTC, 0,
        (long)TT * H1C, (long)H1C * OUTC, (long)TT * OUTC, 0, OUTC);

    // yhats += [inter, inter @ D]
    k_resid<<<(int)(((long)BB * TT * OUTC + 255) / 256), 256, 0, stream>>>(interf, Dmat, out);
}
